// GpTransformer_2972117368892
// MI455X (gfx1250) — matrix operand order repaired and checked by reading, not compiled, not run on hardware
//
#include <hip/hip_runtime.h>
#include <hip/hip_bf16.h>

// ---------------------------------------------------------------------------
// MI455X (gfx1250, wave32) implementation of the GpTransformer reference.
// Dense math runs through v_wmma_f32_16x16x32_f16. GEMM tiles are staged
// global->LDS with global_load_async_to_lds_b128 (ASYNCcnt, double-buffered)
// and converted f32->f16 at fragment-load time. Attention is a flash-style
// online-softmax WMMA kernel with vectorized f32->f16 staging.
//
// Input pointer layout assumption:
//   d_in[0..6]: inp, adj_matrix, mask, tar, enc_padding_mask,
//               look_ahead_mask, dec_padding_mask   (setup_inputs order)
//   d_in[7..]:  params leaves, jax-style flattening (dict keys sorted
//               alphabetically at every level, tuples in order):
//     dec_layers[4] (26 each) | disc (6) | enc_layers[4] (16 each) |
//     final_ln (2) | fusion (18)
// ---------------------------------------------------------------------------

#define BSZ  8
#define SEQ  512     // N == T == 512
#define DM   256
#define NH   8
#define HD   32      // DM / NH
#define DFFN 1024
#define NLAY 4
#define HSL  128     // LSTM hidden

typedef __attribute__((ext_vector_type(16))) _Float16 v16h;
typedef __attribute__((ext_vector_type(8)))  float    v8f;

union FragH { v16h v; _Float16 h[16]; uint32_t u[8]; };

// ---- CDNA5 async global->LDS copy (ASYNCcnt) ------------------------------
// GLOBAL_LOAD_ASYNC_TO_LDS_B128: copies global memory into LDS, 16 bytes
// per lane. VDST holds the per-lane LDS byte offset (wave-relative).
__device__ inline void async_ld16(uint32_t lds_off, const float* g) {
  asm volatile("global_load_async_to_lds_b128 %0, %1, off"
               :: "v"(lds_off), "v"(g) : "memory");
}
__device__ inline void s_wait_async0() {
#if __has_builtin(__builtin_amdgcn_s_wait_asynccnt)
  __builtin_amdgcn_s_wait_asynccnt(0);
#else
  asm volatile("s_wait_asynccnt 0x0" ::: "memory");
#endif
}
__device__ inline void s_wait_async6() {
#if __has_builtin(__builtin_amdgcn_s_wait_asynccnt)
  __builtin_amdgcn_s_wait_asynccnt(6);
#else
  asm volatile("s_wait_asynccnt 0x6" ::: "memory");
#endif
}

// ---- WMMA fragment loaders (CDNA5 16x16x32 f16 layouts, wave32) -----------

// A 16x32 f16 from f16 LDS: lane%16 = row M; lane/16 selects K-halves.
__device__ inline v16h load_a_frag(const _Float16* As, int mrow, int ktile, int stride) {
  int lane = threadIdx.x & 31;
  const _Float16* p = As + (mrow + (lane & 15)) * stride + ktile;
  int kh = ((lane >> 4) & 1) * 16;
  FragH r;
#pragma unroll
  for (int i = 0; i < 8; ++i) r.u[i]     = *(const uint32_t*)(p + kh + 2 * i);
  return r.v;
}

// A 16x32 from f32 LDS (convert on the fly; k-contiguous rows).
__device__ inline v16h load_a_frag_f32(const float* Ss, int mrow, int ktile, int stride) {
  int lane = threadIdx.x & 31;
  const float* p = Ss + (mrow + (lane & 15)) * stride + ktile;
  int kh = ((lane >> 4) & 1) * 16;
  FragH r;
#pragma unroll
  for (int i = 0; i < 16; ++i) r.h[i]    = (_Float16)p[kh + i];
  return r.v;
}

// B 32x16 f16 from f16 LDS stored column-major Bs[n][k].
__device__ inline v16h load_b_frag(const _Float16* Bs, int ncol, int ktile, int stride) {
  int lane = threadIdx.x & 31;
  const _Float16* p = Bs + (ncol + (lane & 15)) * stride + ktile + ((lane >> 4) & 1) * 16;
  FragH r;
#pragma unroll
  for (int i = 0; i < 8; ++i) r.u[i] = *(const uint32_t*)(p + 2 * i);
  return r.v;
}

// B 32x16 from f32 LDS stored [n][k] (k-contiguous -> vectorizable).
__device__ inline v16h b_frag_f32_nk(const float* Bs, int ncol, int stride) {
  int lane = threadIdx.x & 31;
  const float* p = Bs + (ncol + (lane & 15)) * stride + ((lane >> 4) & 1) * 16;
  FragH r;
#pragma unroll
  for (int i = 0; i < 16; ++i) r.h[i] = (_Float16)p[i];
  return r.v;
}

// B 32x16 from f32 LDS stored [k][n] (strided gather; 16 distinct banks).
__device__ inline v16h b_frag_f32_kn(const float* Bs, int ncol, int stride) {
  int lane = threadIdx.x & 31;
  int col = ncol + (lane & 15);
  int kb = ((lane >> 4) & 1) * 16;
  FragH r;
#pragma unroll
  for (int i = 0; i < 16; ++i) r.h[i] = (_Float16)Bs[(kb + i) * stride + col];
  return r.v;
}

__device__ inline v8f wmma_f16(v16h a, v16h b, v8f c) {
  return __builtin_amdgcn_wmma_f32_16x16x32_f16(false, a, false, b, (short)0, c, false, false);
}

// ---------------------------------------------------------------------------
// GEMM: C[bz] = act(A[bz] @ B[bz](^T) + bias + res[bz])
// Block tile 128x64, BK=32; 256 threads (8 waves), each wave a 32x32 tile.
// Double-buffered async global->LDS staging (f32), f16 conversion in frags.
// TRANSB=0: B is (K,N) row-major; TRANSB=1: B is (N,K) row-major.
// All dims here: M %128==0, N %64==0, K %32==0.
// ---------------------------------------------------------------------------
template<int TRANSB>
__global__ __launch_bounds__(256) void gemm_async(
    const float* __restrict__ A, int lda, long long sA,
    const float* __restrict__ B, int ldb, long long sB,
    const float* __restrict__ bias, const float* __restrict__ res,
    float* __restrict__ C, int ldc, long long sC,
    int M, int N, int K, int relu)
{
  constexpr int ASTR  = 36;                 // f32 stride: 144B rows, 16B-aligned
  constexpr int BROWS = TRANSB ? 64 : 32;
  constexpr int BSTR  = TRANSB ? 36 : 68;   // 144B / 272B rows
  __shared__ float As32[2][128 * ASTR];
  __shared__ float Bs32[2][BROWS * BSTR];

  const int bm = blockIdx.y * 128, bn = blockIdx.x * 64;
  const long long bz = blockIdx.z;
  A += bz * sA; B += bz * sB; C += bz * sC;
  const float* R = res ? res + bz * sC : nullptr;

  const int tid = threadIdx.x;
  const int w = tid >> 5, wr = w >> 1, wc = w & 1;

  auto stage = [&](int k0, int buf) {
    uint32_t abase = (uint32_t)(uintptr_t)&As32[buf][0];
#pragma unroll
    for (int i = 0; i < 4; ++i) {           // A: 128 rows x 128B = 1024 chunks
      int c = i * 256 + tid, row = c >> 3, seg = c & 7;
      async_ld16(abase + row * (ASTR * 4) + seg * 16,
                 A + (long long)(bm + row) * lda + k0 + seg * 4);
    }
    uint32_t bbase = (uint32_t)(uintptr_t)&Bs32[buf][0];
    if constexpr (TRANSB) {
#pragma unroll
      for (int i = 0; i < 2; ++i) {         // B: 64 rows x 128B = 512 chunks
        int c = i * 256 + tid, row = c >> 3, seg = c & 7;
        async_ld16(bbase + row * (BSTR * 4) + seg * 16,
                   B + (long long)(bn + row) * ldb + k0 + seg * 4);
      }
    } else {
#pragma unroll
      for (int i = 0; i < 2; ++i) {         // B: 32 rows x 256B = 512 chunks
        int c = i * 256 + tid, row = c >> 4, seg = c & 15;
        async_ld16(bbase + row * (BSTR * 4) + seg * 16,
                   B + (long long)(k0 + row) * ldb + bn + seg * 4);
      }
    }
  };

  v8f acc00 = {}, acc01 = {}, acc10 = {}, acc11 = {};
  const int nk = K >> 5;
  stage(0, 0);
  for (int t = 0; t < nk; ++t) {
    const int cur = t & 1;
    if (t + 1 < nk) {                       // overlap next tile with compute
      stage((t + 1) << 5, cur ^ 1);
      s_wait_async6();                      // first-in-order 6 == tile t done
    } else {
      s_wait_async0();
    }
    __syncthreads();

    const float* Ab = &As32[cur][0];
    const float* Bb = &Bs32[cur][0];
    v16h a0 = load_a_frag_f32(Ab, wr * 32,      0, ASTR);
    v16h a1 = load_a_frag_f32(Ab, wr * 32 + 16, 0, ASTR);
    v16h b0, b1;
    if constexpr (TRANSB) {
      b0 = b_frag_f32_nk(Bb, wc * 32,      BSTR);
      b1 = b_frag_f32_nk(Bb, wc * 32 + 16, BSTR);
    } else {
      b0 = b_frag_f32_kn(Bb, wc * 32,      BSTR);
      b1 = b_frag_f32_kn(Bb, wc * 32 + 16, BSTR);
    }
    acc00 = wmma_f16(a0, b0, acc00);
    acc01 = wmma_f16(a0, b1, acc01);
    acc10 = wmma_f16(a1, b0, acc10);
    acc11 = wmma_f16(a1, b1, acc11);
    __syncthreads();                        // protect buf reuse (t+2 staging)
  }

  // Epilogue. C/D layout: lane%16 = col, rows r + 8*(lane/16).
  const int lane = tid & 31, col = lane & 15, rb = (lane >> 4) * 8;
  v8f* accs[4] = { &acc00, &acc01, &acc10, &acc11 };
#pragma unroll
  for (int t = 0; t < 4; ++t) {
    int i = t >> 1, j = t & 1;
    int gm = bm + wr * 32 + i * 16 + rb;
    int gn = bn + wc * 32 + j * 16 + col;
    float bv = bias ? bias[gn] : 0.f;
#pragma unroll
    for (int r = 0; r < 8; ++r) {
      float v = (*accs[t])[r] + bv;
      if (R) v += R[(long long)(gm + r) * ldc + gn];
      if (relu) v = fmaxf(v, 0.f);
      C[(long long)(gm + r) * ldc + gn] = v;
    }
  }
}

// ---------------------------------------------------------------------------
// Flash attention: one workgroup = (b, h, 64-query block), 256 threads/8 waves.
// logits = (Q*scale) @ K^T + mask*(-1e9); online softmax; O = P @ V.
// Staging uses float4 global reads along the contiguous head-dim axis.
// ---------------------------------------------------------------------------
#define GSTR 48
__global__ __launch_bounds__(256) void attn_flash(
    const float* __restrict__ Q, const float* __restrict__ K,
    const float* __restrict__ V, float* __restrict__ O,
    int Sq, int Sk,
    const float* __restrict__ maskP, int maskBS, int maskQS)
{
  __shared__ _Float16 Qs[64 * GSTR];      // [q][d]  (scale folded in)
  __shared__ _Float16 Ks[64 * GSTR];      // [key][d] == B col-major for Q@K^T
  __shared__ _Float16 Vs[32 * 80];        // [d][key] == B col-major for P@V
  __shared__ float    Ss[64 * 72];        // scores f32, stride 72
  __shared__ float    mrow[64], lrow[64], frow[64];

  const int qb = blockIdx.x * 64, h = blockIdx.y, b = blockIdx.z;
  const int tid = threadIdx.x, w = tid >> 5, lane = tid & 31;
  const int qi = w >> 1, dj = w & 1;          // wave's 16x16 tile of 64x32 out
  const float scale = 0.17677669529663689f;   // 1/sqrt(32)

#pragma unroll
  for (int e = tid; e < 64 * 8; e += 256) {   // Q: 512 x float4, coalesced
    int r = e >> 3, d4 = (e & 7) * 4;
    float4 qv = *(const float4*)&Q[((long long)b * Sq + qb + r) * DM + h * HD + d4];
    _Float16* dst = &Qs[r * GSTR + d4];
    dst[0] = (_Float16)(qv.x * scale);
    dst[1] = (_Float16)(qv.y * scale);
    dst[2] = (_Float16)(qv.z * scale);
    dst[3] = (_Float16)(qv.w * scale);
  }
  if (tid < 64) { mrow[tid] = -1e30f; lrow[tid] = 0.f; }
  v8f acc = {};
  __syncthreads();

  for (int kc = 0; kc < Sk; kc += 64) {
#pragma unroll
    for (int e = tid; e < 64 * 8; e += 256) { // K chunk: float4, coalesced
      int r = e >> 3, d4 = (e & 7) * 4;
      float4 kv = *(const float4*)&K[((long long)b * Sk + kc + r) * DM + h * HD + d4];
      _Float16* dst = &Ks[r * GSTR + d4];
      dst[0] = (_Float16)kv.x;
      dst[1] = (_Float16)kv.y;
      dst[2] = (_Float16)kv.z;
      dst[3] = (_Float16)kv.w;
    }
#pragma unroll
    for (int e = tid; e < 64 * 8; e += 256) { // V chunk: coalesced read,
      int key = e >> 3, d4 = (e & 7) * 4;     // transposed [d][key] LDS store
      float4 vv = *(const float4*)&V[((long long)b * Sk + kc + key) * DM + h * HD + d4];
      Vs[(d4 + 0) * 80 + key] = (_Float16)vv.x;
      Vs[(d4 + 1) * 80 + key] = (_Float16)vv.y;
      Vs[(d4 + 2) * 80 + key] = (_Float16)vv.z;
      Vs[(d4 + 3) * 80 + key] = (_Float16)vv.w;
    }
    __syncthreads();

    for (int t = w; t < 16; t += 8) {       // S = Q @ K^T, 2 tiles per wave
      int ti = t >> 2, tj = t & 3;
      v16h aq = load_a_frag(Qs, ti * 16, 0, GSTR);
      v16h bk = load_b_frag(Ks, tj * 16, 0, GSTR);
      v8f s = {};
      s = wmma_f16(aq, bk, s);
      int cl = lane & 15, rbs = (lane >> 4) * 8;
#pragma unroll
      for (int r = 0; r < 8; ++r)
        Ss[(ti * 16 + rbs + r) * 72 + tj * 16 + cl] = s[r];
    }
    __syncthreads();

    if (tid < 64) {                         // online softmax, 1 thread/row
      int row = tid, gq = qb + row;
      float* sr = &Ss[row * 72];
      float mc = -1e30f;
      for (int j = 0; j < 64; ++j) {
        float mv = maskP ? maskP[(long long)b * maskBS + (long long)gq * maskQS + kc + j] : 0.f;
        float s = sr[j] - 1e9f * mv;
        sr[j] = s;
        mc = fmaxf(mc, s);
      }
      float mo = mrow[row], mn = fmaxf(mo, mc);
      float f = expf(mo - mn);
      float ps = 0.f;
      for (int j = 0; j < 64; ++j) { float p = expf(sr[j] - mn); sr[j] = p; ps += p; }
      mrow[row] = mn; lrow[row] = lrow[row] * f + ps; frow[row] = f;
    }
    __syncthreads();

    {
      int rbs = (lane >> 4) * 8;
#pragma unroll
      for (int r = 0; r < 8; ++r) acc[r] *= frow[qi * 16 + rbs + r];
    }
#pragma unroll
    for (int kk = 0; kk < 64; kk += 32) {   // O += P @ V
      v16h ap = load_a_frag_f32(Ss, qi * 16, kk, 72);
      v16h bv = load_b_frag(Vs, dj * 16, kk, 80);
      acc = wmma_f16(ap, bv, acc);
    }
    __syncthreads();
  }

  {
    int cl = lane & 15, rbs = (lane >> 4) * 8;
#pragma unroll
    for (int r = 0; r < 8; ++r) {
      int row = qi * 16 + rbs + r;
      O[((long long)b * Sq + qb + row) * DM + h * HD + dj * 16 + cl] = acc[r] / lrow[row];
    }
  }
}

// ---------------------------------------------------------------------------
// VALU helper kernels
// ---------------------------------------------------------------------------
__global__ void ln_kernel(const float* __restrict__ x, const float* __restrict__ g,
                          const float* __restrict__ bt, float* __restrict__ y,
                          int rows, int dim) {
  int r = blockIdx.x * blockDim.x + threadIdx.x;
  if (r >= rows) return;
  const float* xr = x + (long long)r * dim;
  float m = 0.f;
  for (int d = 0; d < dim; ++d) m += xr[d];
  m /= dim;
  float v = 0.f;
  for (int d = 0; d < dim; ++d) { float t = xr[d] - m; v += t * t; }
  v /= dim;
  float inv = rsqrtf(v + 1e-6f);
  float* yr = y + (long long)r * dim;
  for (int d = 0; d < dim; ++d) yr[d] = (xr[d] - m) * inv * g[d] + bt[d];
}

__global__ void pe_init(float* pe, int seq, int dim) {
  int i = blockIdx.x * blockDim.x + threadIdx.x;
  if (i >= seq * dim) return;
  int pos = i / dim, d = i % dim;
  float expo = (float)(2 * (d / 2)) / (float)dim;
  float ang = (float)pos / powf(10000.f, expo);
  pe[i] = (d & 1) ? cosf(ang) : sinf(ang);
}

__global__ void add_pe(float* dst, const float* src, const float* pe, int total, int pemod) {
  int i = blockIdx.x * blockDim.x + threadIdx.x;
  if (i >= total) return;
  dst[i] = src[i] + pe[i % pemod];
}

__global__ void mask_mul(float* x, const float* m, int total, int dim) {
  int i = blockIdx.x * blockDim.x + threadIdx.x;
  if (i >= total) return;
  x[i] *= m[i / dim];
}

__global__ void row_sum(const float* a, float* rs, int rows, int cols) {
  int r = blockIdx.x * blockDim.x + threadIdx.x;
  if (r >= rows) return;
  const float* p = a + (long long)r * cols;
  float s = 0.f;
  for (int j = 0; j < cols; ++j) s += p[j];
  rs[r] = s;
}

__global__ void scale_rows(float* x, const float* rs, int total, int dim) {
  int i = blockIdx.x * blockDim.x + threadIdx.x;
  if (i >= total) return;
  x[i] /= (rs[i / dim] + 1e-6f);
}

__global__ void concat3(const float* a, const float* b, const float* c,
                        float* o, int rows) {
  int i = blockIdx.x * blockDim.x + threadIdx.x;
  if (i >= rows * 768) return;
  int r = i / 768, k = i % 768;
  o[i] = (k < 256) ? a[(long long)r * 256 + k]
       : (k < 512) ? b[(long long)r * 256 + (k - 256)]
                   : c[(long long)r * 256 + (k - 512)];
}

__device__ inline float sigm(float x) { return 1.f / (1.f + expf(-x)); }

// Persistent bidirectional-LSTM scan. One block = one (batch, direction).
__global__ __launch_bounds__(128) void lstm_scan(
    const float* __restrict__ XW, const float* __restrict__ Whh,
    float* __restrict__ seqout, int Nlen, int dirBack, int colOff)
{
  int b = blockIdx.x;
  int j = threadIdx.x;
  __shared__ float hs[HSL];
  float c = 0.f;
  hs[j] = 0.f;
  __syncthreads();
  for (int s = 0; s < Nlen; ++s) {
    int t = dirBack ? (Nlen - 1 - s) : s;
    const float* g = &XW[((long long)b * Nlen + t) * (4 * HSL)];
    float gi = g[j], gf = g[HSL + j], gg = g[2 * HSL + j], go = g[3 * HSL + j];
    for (int kx = 0; kx < HSL; ++kx) {
      float hv = hs[kx];
      const float* wr = &Whh[(long long)kx * (4 * HSL)];
      gi += hv * wr[j];
      gf += hv * wr[HSL + j];
      gg += hv * wr[2 * HSL + j];
      go += hv * wr[3 * HSL + j];
    }
    c = sigm(gf) * c + sigm(gi) * tanhf(gg);
    float hn = sigm(go) * tanhf(c);
    __syncthreads();
    hs[j] = hn;
    seqout[((long long)b * Nlen + t) * (2 * HSL) + colOff + j] = hn;
    __syncthreads();
  }
}

// ---------------------------------------------------------------------------
// Host orchestration
// ---------------------------------------------------------------------------
extern "C" void kernel_launch(void* const* d_in, const int* in_sizes, int n_in,
                              void* d_out, int out_size, void* d_ws, size_t ws_size,
                              hipStream_t stream) {
  (void)in_sizes; (void)n_in; (void)out_size; (void)ws_size;

  const float* inp     = (const float*)d_in[0];
  const float* adj     = (const float*)d_in[1];
  const float* maskv   = (const float*)d_in[2];
  const float* tar     = (const float*)d_in[3];
  const float* encmask = (const float*)d_in[4];
  const float* lam     = (const float*)d_in[5];
  const float* decmask = (const float*)d_in[6];
  auto P = [&](int i) { return (const float*)d_in[7 + i]; };

  const int DEC0 = 0;        // + l*26
  const int DISC = 104;
  const int ENC0 = 110;      // + l*16
  const int FLN  = 174;
  const int FUS  = 176;

  float* Wb = (float*)d_ws;
  size_t off = 0;
  auto alloc = [&](size_t n) { float* p = Wb + off; off += n; return p; };
  const int RB = BSZ * SEQ;                 // 4096 rows
  float* PEb  = alloc((size_t)SEQ * DM);
  float* tmpA = alloc((size_t)RB * 1024);
  float* tmpB = alloc((size_t)RB * 1024);
  float* tmpC = alloc((size_t)RB * DM);
  float* seqb = alloc((size_t)RB * DM);
  float* seqf = alloc((size_t)RB * DM);
  float* spa1 = alloc((size_t)RB * HSL);
  float* spa2 = alloc((size_t)RB * HSL);
  float* spap = alloc((size_t)RB * DM);
  float* rs   = alloc((size_t)BSZ * SEQ);
  float* xA   = alloc((size_t)RB * DM);
  float* xB   = alloc((size_t)RB * DM);
  float* yA   = alloc((size_t)RB * DM);
  float* yB   = alloc((size_t)RB * DM);
  float* lnb  = alloc((size_t)RB * DM);
  float* Qb   = alloc((size_t)RB * DM);
  float* Kb   = alloc((size_t)RB * DM);
  float* Vb   = alloc((size_t)RB * DM);
  float* aO   = alloc((size_t)RB * DM);
  float* ffh  = alloc((size_t)RB * DFFN);
  float* dpj  = alloc((size_t)RB * DM);

  auto gemm = [&](const float* A, int lda, long long sA,
                  const float* B, int ldb, long long sB, int tB,
                  const float* bias, const float* res,
                  float* C, int ldc, long long sC,
                  int M, int N, int K, int relu, int batch) {
    dim3 g(N / 64, M / 128, batch);
    if (tB)
      gemm_async<1><<<g, dim3(256), 0, stream>>>(A, lda, sA, B, ldb, sB,
                                                 bias, res, C, ldc, sC, M, N, K, relu);
    else
      gemm_async<0><<<g, dim3(256), 0, stream>>>(A, lda, sA, B, ldb, sB,
                                                 bias, res, C, ldc, sC, M, N, K, relu);
  };
  auto ew = [&](int n) { return dim3((n + 255) / 256); };
  auto ln = [&](const float* x, const float* g, const float* b, float* y) {
    ln_kernel<<<ew(RB), 256, 0, stream>>>(x, g, b, y, RB, DM);
  };
  auto attn = [&](const float* Qp, const float* Kp, const float* Vp, float* Op,
                  const float* mp, int mbs, int mqs) {
    attn_flash<<<dim3(SEQ / 64, NH, BSZ), 256, 0, stream>>>(Qp, Kp, Vp, Op, SEQ, SEQ, mp, mbs, mqs);
  };

  // ---- positional encoding -------------------------------------------------
  pe_init<<<ew(SEQ * DM), 256, 0, stream>>>(PEb, SEQ, DM);

  // ---- fusion: BiLSTM ------------------------------------------------------
  gemm(inp, DM, 0, P(FUS + 9), 4 * HSL, 0, 0, P(FUS + 11), nullptr,
       tmpA, 4 * HSL, 0, RB, 4 * HSL, DM, 0, 1);                       // x@Wih_f+b_f
  gemm(inp, DM, 0, P(FUS + 8), 4 * HSL, 0, 0, P(FUS + 10), nullptr,
       tmpB, 4 * HSL, 0, RB, 4 * HSL, DM, 0, 1);                       // x@Wih_b+b_b
  lstm_scan<<<BSZ, HSL, 0, stream>>>(tmpA, P(FUS + 7), seqb, SEQ, 0, 0);
  lstm_scan<<<BSZ, HSL, 0, stream>>>(tmpB, P(FUS + 6), seqb, SEQ, 1, HSL);
  gemm(seqb, DM, 0, P(FUS + 16), DM, 0, 0, P(FUS + 17), nullptr,
       seqf, DM, 0, RB, DM, DM, 0, 1);                                 // seq_proj

  // ---- fusion: GCN ---------------------------------------------------------
  row_sum<<<ew(BSZ * SEQ), 256, 0, stream>>>(adj, rs, BSZ * SEQ, SEQ);
  gemm(adj, SEQ, (long long)SEQ * SEQ, inp, DM, (long long)SEQ * DM, 0,
       nullptr, nullptr, tmpA, DM, (long long)SEQ * DM, SEQ, DM, SEQ, 0, BSZ);
  scale_rows<<<ew(RB * DM), 256, 0, stream>>>(tmpA, rs, RB * DM, DM);
  gemm(tmpA, DM, 0, P(FUS + 0), HSL, 0, 0, P(FUS + 1), nullptr,
       spa1, HSL, 0, RB, HSL, DM, 1, 1);                               // gcn0 + relu
  gemm(adj, SEQ, (long long)SEQ * SEQ, spa1, HSL, (long long)SEQ * HSL, 0,
       nullptr, nullptr, tmpB, HSL, (long long)SEQ * HSL, SEQ, HSL, SEQ, 0, BSZ);
  scale_rows<<<ew(RB * HSL), 256, 0, stream>>>(tmpB, rs, RB * HSL, HSL);
  gemm(tmpB, HSL, 0, P(FUS + 2), HSL, 0, 0, P(FUS + 3), spa1,
       spa2, HSL, 0, RB, HSL, HSL, 1, 1);                              // gcn1 + res + relu
  gemm(spa2, HSL, 0, P(FUS + 4), DM, 0, 0, P(FUS + 5), nullptr,
       spap, DM, 0, RB, DM, HSL, 0, 1);                                // gcn_proj

  // ---- fusion: concat + MLP + mask + PE -----------------------------------
  concat3<<<ew(RB * 768), 256, 0, stream>>>(inp, seqf, spap, tmpA, RB);
  gemm(tmpA, 768, 0, P(FUS + 12), DM, 0, 0, P(FUS + 13), nullptr,
       tmpC, DM, 0, RB, DM, 768, 1, 1);                                // mlp1 + relu
  gemm(tmpC, DM, 0, P(FUS + 14), DM, 0, 0, P(FUS + 15), nullptr,
       xA, DM, 0, RB, DM, DM, 0, 1);                                   // mlp2
  mask_mul<<<ew(RB * DM), 256, 0, stream>>>(xA, maskv, RB * DM, DM);
  add_pe<<<ew(RB * DM), 256, 0, stream>>>(xA, xA, PEb, RB * DM, SEQ * DM);

  // ---- encoder -------------------------------------------------------------
  float* x = xA; float* xo = xB;
  for (int l = 0; l < NLAY; ++l) {
    const int E = ENC0 + l * 16;
    ln(x, P(E + 4), P(E + 5), lnb);
    gemm(lnb, DM, 0, P(E + 12), DM, 0, 0, P(E + 13), nullptr, Qb, DM, 0, RB, DM, DM, 0, 1);
    gemm(lnb, DM, 0, P(E + 10), DM, 0, 0, P(E + 11), nullptr, Kb, DM, 0, RB, DM, DM, 0, 1);
    gemm(lnb, DM, 0, P(E + 14), DM, 0, 0, P(E + 15), nullptr, Vb, DM, 0, RB, DM, DM, 0, 1);
    attn(Qb, Kb, Vb, aO, encmask, SEQ, 0);
    gemm(aO, DM, 0, P(E + 8), DM, 0, 0, P(E + 9), x, xo, DM, 0, RB, DM, DM, 0, 1);
    { float* t = x; x = xo; xo = t; }
    ln(x, P(E + 6), P(E + 7), lnb);
    gemm(lnb, DM, 0, P(E + 0), DFFN, 0, 0, P(E + 1), nullptr, ffh, DFFN, 0, RB, DFFN, DM, 1, 1);
    gemm(ffh, DFFN, 0, P(E + 2), DM, 0, 0, P(E + 3), x, xo, DM, 0, RB, DM, DFFN, 0, 1);
    { float* t = x; x = xo; xo = t; }
  }
  const float* enc_out = x;

  // ---- decoder -------------------------------------------------------------
  add_pe<<<ew(RB * DM), 256, 0, stream>>>(yA, tar, PEb, RB * DM, SEQ * DM);
  float* y = yA; float* yo = yB;
  for (int l = 0; l < NLAY; ++l) {
    const int D0 = DEC0 + l * 26;
    // self-attention (look-ahead mask)
    ln(y, P(D0 + 4), P(D0 + 5), lnb);
    gemm(lnb, DM, 0, P(D0 + 14), DM, 0, 0, P(D0 + 15), nullptr, Qb, DM, 0, RB, DM, DM, 0, 1);
    gemm(lnb, DM, 0, P(D0 + 12), DM, 0, 0, P(D0 + 13), nullptr, Kb, DM, 0, RB, DM, DM, 0, 1);
    gemm(lnb, DM, 0, P(D0 + 16), DM, 0, 0, P(D0 + 17), nullptr, Vb, DM, 0, RB, DM, DM, 0, 1);
    attn(Qb, Kb, Vb, aO, lam, 0, SEQ);
    gemm(aO, DM, 0, P(D0 + 10), DM, 0, 0, P(D0 + 11), y, yo, DM, 0, RB, DM, DM, 0, 1);
    { float* t = y; y = yo; yo = t; }
    // cross-attention (q from decoder, k/v from encoder output)
    ln(y, P(D0 + 6), P(D0 + 7), lnb);
    gemm(lnb,     DM, 0, P(D0 + 22), DM, 0, 0, P(D0 + 23), nullptr, Qb, DM, 0, RB, DM, DM, 0, 1);
    gemm(enc_out, DM, 0, P(D0 + 20), DM, 0, 0, P(D0 + 21), nullptr, Kb, DM, 0, RB, DM, DM, 0, 1);
    gemm(enc_out, DM, 0, P(D0 + 24), DM, 0, 0, P(D0 + 25), nullptr, Vb, DM, 0, RB, DM, DM, 0, 1);
    attn(Qb, Kb, Vb, aO, decmask, SEQ, 0);
    gemm(aO, DM, 0, P(D0 + 18), DM, 0, 0, P(D0 + 19), y, yo, DM, 0, RB, DM, DM, 0, 1);
    { float* t = y; y = yo; yo = t; }
    // FFN
    ln(y, P(D0 + 8), P(D0 + 9), lnb);
    gemm(lnb, DM, 0, P(D0 + 0), DFFN, 0, 0, P(D0 + 1), nullptr, ffh, DFFN, 0, RB, DFFN, DM, 1, 1);
    gemm(ffh, DFFN, 0, P(D0 + 2), DM, 0, 0, P(D0 + 3), y, yo, DM, 0, RB, DM, DFFN, 0, 1);
    { float* t = y; y = yo; yo = t; }
  }

  // ---- final LN + discriminator head + output einsum ----------------------
  ln(y, P(FLN + 0), P(FLN + 1), lnb);
  gemm(lnb,  DM,   0, P(DISC + 0), 1024, 0, 0, P(DISC + 1), nullptr, tmpA, 1024, 0, RB, 1024, DM,   0, 1);
  gemm(tmpA, 1024, 0, P(DISC + 2), 512,  0, 0, P(DISC + 3), nullptr, tmpB, 512,  0, RB, 512,  1024, 0, 1);
  gemm(tmpB, 512,  0, P(DISC + 4), DM,   0, 0, P(DISC + 5), nullptr, dpj,  DM,   0, RB, DM,   512,  0, 1);

  // out[b,g,s] = sum_d dproj[b,g,d] * enc_out[b,s,d]  ->  batched GEMM, B (N,K)
  gemm(dpj, DM, (long long)SEQ * DM, enc_out, DM, (long long)SEQ * DM, 1,
       nullptr, nullptr, (float*)d_out, SEQ, (long long)SEQ * SEQ,
       SEQ, SEQ, DM, 0, BSZ);
}
